// LHC_63806034150143
// MI455X (gfx1250) — matrix operand order repaired and checked by reading, not compiled, not run on hardware
//
#include <hip/hip_runtime.h>
#include <math.h>

typedef __attribute__((ext_vector_type(16))) _Float16 v16h;
typedef __attribute__((ext_vector_type(8)))  _Float16 v8h;
typedef __attribute__((ext_vector_type(8)))  float    v8f;
typedef _Float16 h16;

#define WMMA16(a,b,c) __builtin_amdgcn_wmma_f32_16x16x32_f16(false,(a),false,(b),(short)0,(c),false,false)

#define NFR   48
#define FGEN  47      // generated frames
#define NPOS  1024    // 32*32 latent positions

__device__ __forceinline__ int reflect1(int v, int n) {
  v = (v < 0) ? -v : v;                   // pad=1 'reflect': -1 -> 1
  return (v >= n) ? (2 * n - 2 - v) : v;  // n -> n-2
}
__device__ __forceinline__ int clampi(int v, int lo, int hi) {
  v = (v < lo) ? lo : v;
  return (v > hi) ? hi : v;
}
__device__ __forceinline__ float lin32(int i) {
  return -1.0f + (2.0f / 31.0f) * (float)i;
}
// contiguous 16-half run (k = 2*kbase .. 2*kbase+15) as an A fragment
__device__ __forceinline__ v16h load_a16(const h16* p, int kbase) {
  v16h a = *(const v16h*)(p + 2 * kbase);
  return a;
}

// ---------------------------------------------------------------------------
// Prep kernels
// ---------------------------------------------------------------------------
// weights (Cout,Cin,3,3) f32 -> f16 transposed to [tap][cout][cin]
__global__ void cvt_w_tr(const float* __restrict__ s, h16* __restrict__ d,
                         int COUT, int CIN, int n) {
  int i = blockIdx.x * blockDim.x + threadIdx.x;
  if (i >= n) return;
  int co = i / (CIN * 9);
  int r = i - co * (CIN * 9);
  int ci = r / 9, tap = r % 9;
  d[(tap * COUT + co) * CIN + ci] = (h16)s[i];
}

// xs = (x*2-1) as f16 NHWC; frame 0 of output == x (normalizations cancel)
__global__ void prep_input(const float* __restrict__ x, h16* __restrict__ xs,
                           float* __restrict__ dout) {
  int i = blockIdx.x * blockDim.x + threadIdx.x;      // 4*3*16384
  if (i >= 4 * 3 * 16384) return;
  float v = x[i];
  int b = i / 49152;
  int r = i - b * 49152;
  int c = r / 16384, p = r - c * 16384;
  xs[((size_t)b * 16384 + p) * 3 + c] = (h16)(v * 2.0f - 1.0f);
  dout[(size_t)b * NFR * 49152 + r] = v;
}

// kdT[p][n] = relu(1 - dist(pos_n, grid_p)/sqrt(2048)) (transposed, f16)
__global__ void build_kdist(h16* __restrict__ kdT) {
  int i = blockIdx.x * blockDim.x + threadIdx.x;      // 1024*1024
  if (i >= NPOS * NPOS) return;
  int p = i >> 10, n = i & 1023;
  float nx = lin32(n >> 5), ny = lin32(n & 31);
  float gx = lin32(p >> 5), gy = lin32(p & 31);
  float dx = nx - gx, dy = ny - gy;
  float v = 1.0f - 0.02209708691207961f * (dx * dx + dy * dy);
  kdT[i] = (h16)fmaxf(v, 0.0f);
}

// knn[n][0..7] = indices of 8 smallest gram values (rank-2 gram), stable ties
__global__ void build_knn(int* __restrict__ knn) {
  int i = blockIdx.x * blockDim.x + threadIdx.x;
  if (i >= NPOS) return;
  float px = lin32(i >> 5), py = lin32(i & 31);
  float bv[8]; int bi[8];
  #pragma unroll
  for (int t = 0; t < 8; ++t) { bv[t] = 1e30f; bi[t] = 0; }
  for (int j = 0; j < NPOS; ++j) {
    float qx = lin32(j >> 5), qy = lin32(j & 31);
    float g = px * qx + py * qy;
    if (g < bv[7]) {
      int t = 7;
      while (t > 0 && g < bv[t - 1]) { bv[t] = bv[t - 1]; bi[t] = bi[t - 1]; --t; }
      bv[t] = g; bi[t] = j;
    }
  }
  #pragma unroll
  for (int t = 0; t < 8; ++t) knn[i * 8 + t] = bi[t];
}

// xp_next[bc][n] = mean_{j in knn[n]} xp[bc][j]   (channel-major (B,C,N))
__global__ void knn_gather(const h16* __restrict__ xin, const int* __restrict__ knn,
                           h16* __restrict__ xout) {
  int i = blockIdx.x * blockDim.x + threadIdx.x;      // 128*1024
  if (i >= 128 * NPOS) return;
  int bc = i >> 10, n = i & 1023;
  const int* kk = knn + n * 8;
  float s = 0.0f;
  #pragma unroll
  for (int j = 0; j < 8; ++j) s += (float)xin[bc * NPOS + kk[j]];
  xout[i] = (h16)(s * 0.125f);
}

// 2x2 average pool on NHWC
__global__ void avgpool2_nhwc(const h16* __restrict__ in, h16* __restrict__ out,
                              int outRes, int C, int total) {
  int i = blockIdx.x * blockDim.x + threadIdx.x;      // nImg*Pout*C
  if (i >= total) return;
  int c = i % C;
  int j = i / C;
  int Pout = outRes * outRes;
  int op = j % Pout, img = j / Pout;
  int oy = op / outRes, ox = op % outRes;
  int inRes = outRes * 2;
  const h16* b = in + ((size_t)img * inRes * inRes + (2 * oy) * inRes + 2 * ox) * C + c;
  float s = (float)b[0] + (float)b[C] + (float)b[inRes * C] + (float)b[(inRes + 1) * C];
  out[i] = (h16)(s * 0.25f);
}

// 2x2 average pool NHWC (4 imgs, 64->32, C=32) into channel-major (B,C,N)
__global__ void avgpool2_cmajor(const h16* __restrict__ in, h16* __restrict__ out) {
  int i = blockIdx.x * blockDim.x + threadIdx.x;      // 4*32*1024
  if (i >= 4 * 32 * NPOS) return;
  int op = i & 1023;
  int r = i >> 10;
  int c = r & 31, img = r >> 5;
  int oy = op >> 5, ox = op & 31;
  const h16* b = in + (((size_t)img * 4096 + (2 * oy) * 64 + 2 * ox) << 5) + c;
  float s = (float)b[0] + (float)b[32] + (float)b[64 * 32] + (float)b[65 * 32];
  out[((size_t)img * 32 + c) * NPOS + op] = (h16)(s * 0.25f);
}

// ---------------------------------------------------------------------------
// WMMA implicit-GEMM 3x3 conv + bias + ReLU, NHWC activations, weights staged
// in LDS as [tap][co][ci].  One wave = 32 cout x 32 pos tile (2x2 WMMA block).
// ---------------------------------------------------------------------------
template <int CIN, int COUT, bool REFLECT, bool UPS>
__global__ __launch_bounds__(128)
void conv_wmma_relu(const h16* __restrict__ in, const h16* __restrict__ whT,
                    const float* __restrict__ bias, h16* __restrict__ out,
                    int res, int tilesPerImg, int totalTiles) {
  __shared__ h16 wlds[9 * COUT * CIN];
  {
    v8h* dv = (v8h*)wlds;
    const v8h* sv = (const v8h*)whT;
    const int tot8 = 9 * COUT * CIN / 8;
    for (int i = threadIdx.x; i < tot8; i += blockDim.x) dv[i] = sv[i];
  }
  __syncthreads();

  int wave = blockIdx.x * (blockDim.x >> 5) + (threadIdx.x >> 5);
  if (wave >= totalTiles) return;
  int lane = threadIdx.x & 31;
  int img = wave / tilesPerImg;
  int t = wave - img * tilesPerImg;
  int P = res * res;
  int nblk = P >> 5;                       // 32-position blocks (rows % 32 == 0)
  int co0 = (t / nblk) << 5;
  int p0 = (t % nblk) << 5;
  int oy = p0 / res, ox0 = p0 % res;

  int n = lane & 15;
  int kgrp = (lane < 16) ? 0 : 16;
  int kbase = (lane < 16) ? 0 : 8;

  const int inRes = UPS ? (res >> 1) : res;
  const h16* inImg = in + (size_t)img * CIN * inRes * inRes;
  const h16* wr0base = wlds + (size_t)(co0 + n) * CIN;       // rows co0+n
  const h16* wr1base = wr0base + 16 * CIN;                   // rows co0+16+n

  v8f acc00 = {}, acc01 = {}, acc10 = {}, acc11 = {};
  #pragma unroll
  for (int tap = 0; tap < 9; ++tap) {
    int ty = tap / 3 - 1, tx = tap % 3 - 1;
    int yy = oy + ty;
    int xxA = ox0 + n + tx;
    int xxB = xxA + 16;
    bool zA = false, zB = false;
    int yz, xAz, xBz;
    if (REFLECT) {
      yz = reflect1(yy, res); xAz = reflect1(xxA, res); xBz = reflect1(xxB, res);
    } else {
      bool yOOB = (yy < 0) || (yy >= res);
      zA = yOOB || (xxA < 0) || (xxA >= res);
      zB = yOOB || (xxB >= res);
      yz = clampi(yy, 0, res - 1); xAz = clampi(xxA, 0, res - 1); xBz = clampi(xxB, 0, res - 1);
    }
    if (UPS) { yz >>= 1; xAz >>= 1; xBz >>= 1; }
    const h16* pixA = inImg + (size_t)(yz * inRes + xAz) * CIN;
    const h16* pixB = inImg + (size_t)(yz * inRes + xBz) * CIN;
    const h16* wrow0 = wr0base + (size_t)tap * COUT * CIN;
    const h16* wrow1 = wr1base + (size_t)tap * COUT * CIN;
    for (int ci0 = 0; ci0 < CIN; ci0 += 32) {
      v16h A0 = load_a16(wrow0 + ci0, kbase);   // ds_load_b128 x2
      v16h A1 = load_a16(wrow1 + ci0, kbase);
      v16h B0 = zA ? (v16h)(h16)0.0f : *(const v16h*)(pixA + ci0 + kgrp);
      v16h B1 = zB ? (v16h)(h16)0.0f : *(const v16h*)(pixB + ci0 + kgrp);
      acc00 = WMMA16(A0, B0, acc00);
      acc01 = WMMA16(A0, B1, acc01);
      acc10 = WMMA16(A1, B0, acc10);
      acc11 = WMMA16(A1, B1, acc11);
    }
  }
  int mb = (lane < 16) ? 0 : 8;
  size_t colA = ((size_t)img * P + p0 + n) * COUT;
  size_t colB = colA + (size_t)16 * COUT;
  v8h o00, o01, o10, o11;
  #pragma unroll
  for (int r = 0; r < 8; ++r) {
    float b0 = bias[co0 + mb + r], b1 = bias[co0 + 16 + mb + r];
    o00[r] = (h16)fmaxf(acc00[r] + b0, 0.0f);
    o01[r] = (h16)fmaxf(acc01[r] + b0, 0.0f);
    o10[r] = (h16)fmaxf(acc10[r] + b1, 0.0f);
    o11[r] = (h16)fmaxf(acc11[r] + b1, 0.0f);
  }
  *(v8h*)(out + colA + co0 + mb)      = o00;
  *(v8h*)(out + colA + co0 + 16 + mb) = o10;
  *(v8h*)(out + colB + co0 + mb)      = o01;
  *(v8h*)(out + colB + co0 + 16 + mb) = o11;
}

// Encoder conv1: Cin=3, all 9 taps packed into one K=32 chunk (k = tap*3 + ci)
__global__ __launch_bounds__(128)
void conv1_wmma(const h16* __restrict__ in, const h16* __restrict__ whT,
                const float* __restrict__ bias, h16* __restrict__ out,
                int totalTiles) {
  const int res = 128, P = 16384, ntiles = 1024, COUT = 32;
  const int tilesPerImg = 2 * ntiles;
  int wave = blockIdx.x * (blockDim.x >> 5) + (threadIdx.x >> 5);
  if (wave >= totalTiles) return;
  int lane = threadIdx.x & 31;
  int img = wave / tilesPerImg;
  int t = wave - img * tilesPerImg;
  int co0 = (t / ntiles) << 4;
  int p0 = (t % ntiles) << 4;
  int oy = p0 / res, ox0 = p0 % res;
  int n = lane & 15;
  int kgrp = (lane < 16) ? 0 : 16;
  int kbase = (lane < 16) ? 0 : 8;
  int co = co0 + n;
  const h16* inImg = in + (size_t)img * 3 * P;

  v16h a, b;
  #pragma unroll
  for (int h = 0; h < 16; ++h) {
    int ka = kgrp + h;
    a[h] = (ka < 27) ? whT[((ka / 3) * COUT + co) * 3 + (ka % 3)] : (h16)0.0f;
    int kb = kgrp + h;
    if (kb < 27) {
      int tap = kb / 3, ci = kb % 3;
      int yy = reflect1(oy + tap / 3 - 1, res);
      int xx = reflect1(ox0 + n + tap % 3 - 1, res);
      b[h] = inImg[(size_t)(yy * res + xx) * 3 + ci];
    } else b[h] = (h16)0.0f;
  }
  v8f acc = {};
  acc = WMMA16(a, b, acc);

  int mb = (lane < 16) ? 0 : 8;
  int com0 = co0 + mb;
  v8h o;
  #pragma unroll
  for (int r = 0; r < 8; ++r) o[r] = (h16)fmaxf(acc[r] + bias[com0 + r], 0.0f);
  *(v8h*)(out + ((size_t)img * P + p0 + n) * COUT + com0) = o;
}

// Frame einsum: y_f[(b,c),p] = sum_n xp[(b,c),n]*kdist[n,p]  (M=128,K=1024,N=1024)
// 2x2 register blocking: one wave = 32 rows x 32 cols, 4 accumulators.
__global__ __launch_bounds__(128)
void gemm_frame_wmma(const h16* __restrict__ xp, const h16* __restrict__ kdT,
                     h16* __restrict__ y, int f) {
  int wave = blockIdx.x * (blockDim.x >> 5) + (threadIdx.x >> 5);
  if (wave >= 128) return;                 // 4 M-blocks * 32 N-blocks
  int lane = threadIdx.x & 31;
  int m0 = (wave >> 5) << 5;
  int p0 = (wave & 31) << 5;
  int n = lane & 15;
  int kgrp = (lane < 16) ? 0 : 16;
  int kbase = (lane < 16) ? 0 : 8;
  const h16* arow0 = xp + (size_t)(m0 + n) * NPOS;
  const h16* arow1 = arow0 + (size_t)16 * NPOS;
  const h16* brow0 = kdT + (size_t)(p0 + n) * NPOS;
  const h16* brow1 = brow0 + (size_t)16 * NPOS;

  v8f acc00 = {}, acc01 = {}, acc10 = {}, acc11 = {};
  for (int k0 = 0; k0 < NPOS; k0 += 32) {
    __builtin_prefetch(brow0 + k0 + 512, 0, 1);   // global_prefetch
    __builtin_prefetch(brow1 + k0 + 512, 0, 1);
    v16h A0 = load_a16(arow0 + k0, kbase);
    v16h A1 = load_a16(arow1 + k0, kbase);
    v16h B0 = *(const v16h*)(brow0 + k0 + kgrp);
    v16h B1 = *(const v16h*)(brow1 + k0 + kgrp);
    acc00 = WMMA16(A0, B0, acc00);
    acc01 = WMMA16(A0, B1, acc01);
    acc10 = WMMA16(A1, B0, acc10);
    acc11 = WMMA16(A1, B1, acc11);
  }
  int mb = (lane < 16) ? 0 : 8;
  int bb = m0 >> 5;                 // both row-subtiles stay in one batch image
  int c0 = mb, c1 = 16 + mb;        // channel offsets within the 32-ch image
  size_t imgBase = (size_t)(bb * FGEN + f) * NPOS;
  size_t colA = (imgBase + p0 + n) * 32;
  size_t colB = colA + 16 * 32;
  v8h o00, o01, o10, o11;
  #pragma unroll
  for (int r = 0; r < 8; ++r) {
    o00[r] = (h16)acc00[r]; o01[r] = (h16)acc01[r];
    o10[r] = (h16)acc10[r]; o11[r] = (h16)acc11[r];
  }
  *(v8h*)(y + colA + c0) = o00;
  *(v8h*)(y + colA + c1) = o10;
  *(v8h*)(y + colB + c0) = o01;
  *(v8h*)(y + colB + c1) = o11;
}

// Decoder conv3: Cin=32, Cout=3, fused 2x upsample of 64-res NHWC input,
// reflect @128, (tanh(v)+1)/2 -> f32 d_out at frame f+1.  32 positions/wave.
__global__ __launch_bounds__(128)
void conv_dec3_wmma(const h16* __restrict__ in, const h16* __restrict__ whT,
                    const float* __restrict__ bias, float* __restrict__ dout,
                    int baseImg, int totalTiles) {
  const int res = 128, P = 16384, inRes = 64, nblk = 512;
  int wave = blockIdx.x * (blockDim.x >> 5) + (threadIdx.x >> 5);
  if (wave >= totalTiles) return;
  int lane = threadIdx.x & 31;
  int li = wave / nblk;
  int p0 = (wave - li * nblk) << 5;
  int img = baseImg + li;
  int oy = p0 / res, ox0 = p0 % res;
  int n = lane & 15;
  int kgrp = (lane < 16) ? 0 : 16;
  int kbase = (lane < 16) ? 0 : 8;
  int m = lane & 15;
  const h16* inImg = in + (size_t)li * 32 * inRes * inRes;

  v8f acc0 = {}, acc1 = {};
  #pragma unroll
  for (int tap = 0; tap < 9; ++tap) {
    int yy = reflect1(oy + tap / 3 - 1, res) >> 1;
    int xA = reflect1(ox0 + n + tap % 3 - 1, res) >> 1;
    int xB = reflect1(ox0 + 16 + n + tap % 3 - 1, res) >> 1;
    v16h a;
    if (m < 3) a = load_a16(whT + (size_t)(tap * 3 + m) * 32, kbase);
    else       a = (v16h)(h16)0.0f;
    v16h B0 = *(const v16h*)(inImg + (size_t)(yy * inRes + xA) * 32 + kgrp);
    v16h B1 = *(const v16h*)(inImg + (size_t)(yy * inRes + xB) * 32 + kgrp);
    acc0 = WMMA16(a, B0, acc0);
    acc1 = WMMA16(a, B1, acc1);
  }
  int mb = (lane < 16) ? 0 : 8;
  int b_ = img / FGEN, f = img % FGEN;
  size_t base = ((size_t)b_ * NFR + (f + 1)) * 3 * (size_t)P;
  #pragma unroll
  for (int r = 0; r < 8; ++r) {
    int com = mb + r;
    if (com < 3) {
      float v0 = tanhf(acc0[r] + bias[com]);
      float v1 = tanhf(acc1[r] + bias[com]);
      dout[base + (size_t)com * P + p0 + n]      = (v0 + 1.0f) * 0.5f;
      dout[base + (size_t)com * P + p0 + 16 + n] = (v1 + 1.0f) * 0.5f;
    }
  }
}

// ---------------------------------------------------------------------------
extern "C" void kernel_launch(void* const* d_in, const int* in_sizes, int n_in,
                              void* d_out, int out_size, void* d_ws, size_t ws_size,
                              hipStream_t stream) {
  const float* x   = (const float*)d_in[0];
  const float* ew1 = (const float*)d_in[1];  const float* eb1 = (const float*)d_in[2];
  const float* ew2 = (const float*)d_in[3];  const float* eb2 = (const float*)d_in[4];
  const float* ew3 = (const float*)d_in[5];  const float* eb3 = (const float*)d_in[6];
  const float* dw1 = (const float*)d_in[7];  const float* db1 = (const float*)d_in[8];
  const float* dw2 = (const float*)d_in[9];  const float* db2 = (const float*)d_in[10];
  const float* dw3 = (const float*)d_in[11]; const float* db3 = (const float*)d_in[12];
  float* out = (float*)d_out;

  char* wsb = (char*)d_ws;
  size_t off = 0;
  auto alloc = [&](size_t bytes) -> void* {
    void* p = wsb + off;
    off = (off + bytes + 255) & ~(size_t)255;
    return p;
  };
  h16* whE1 = (h16*)alloc(864 * 2);
  h16* whE2 = (h16*)alloc(18432 * 2);
  h16* whE3 = (h16*)alloc(18432 * 2);
  h16* whD1 = (h16*)alloc(18432 * 2);
  h16* whD2 = (h16*)alloc(18432 * 2);
  h16* whD3 = (h16*)alloc(864 * 2);
  h16* xs   = (h16*)alloc((size_t)196608 * 2);          // NHWC (4,16384,3)
  h16* h1   = (h16*)alloc((size_t)2097152 * 2);         // NHWC (4,16384,32)
  h16* t2   = (h16*)alloc((size_t)4194304 * 2);         // NHWC (4,16384,64)
  h16* h2   = (h16*)alloc((size_t)1048576 * 2);         // NHWC (4,4096,64)
  h16* t3   = (h16*)alloc((size_t)524288 * 2);          // NHWC (4,4096,32)
  h16* h3   = (h16*)alloc((size_t)131072 * 2);          // cmajor (128,1024) = xp0
  h16* xpA  = (h16*)alloc((size_t)131072 * 2);
  h16* xpB  = (h16*)alloc((size_t)131072 * 2);
  h16* kdT  = (h16*)alloc((size_t)NPOS * NPOS * 2);     // transposed [p][k]
  int* knn  = (int*)alloc((size_t)NPOS * 8 * 4);
  h16* ybuf = (h16*)alloc((size_t)188 * NPOS * 32 * 2); // NHWC (188,1024,32)
  const int CHUNK = 47;                                  // 188 = 4*47
  h16* a1   = (h16*)alloc((size_t)CHUNK * 4096 * 64 * 2);
  h16* a2   = (h16*)alloc((size_t)CHUNK * 4096 * 32 * 2);

  // -- prep: weight transpose+convert, input normalize + frame-0 copy, geometry --
  cvt_w_tr<<<(864 + 255) / 256, 256, 0, stream>>>(ew1, whE1, 32, 3, 864);
  cvt_w_tr<<<(18432 + 255) / 256, 256, 0, stream>>>(ew2, whE2, 64, 32, 18432);
  cvt_w_tr<<<(18432 + 255) / 256, 256, 0, stream>>>(ew3, whE3, 32, 64, 18432);
  cvt_w_tr<<<(18432 + 255) / 256, 256, 0, stream>>>(dw1, whD1, 64, 32, 18432);
  cvt_w_tr<<<(18432 + 255) / 256, 256, 0, stream>>>(dw2, whD2, 32, 64, 18432);
  cvt_w_tr<<<(864 + 255) / 256, 256, 0, stream>>>(dw3, whD3, 3, 32, 864);
  prep_input<<<768, 256, 0, stream>>>(x, xs, out);
  build_kdist<<<4096, 256, 0, stream>>>(kdT);
  build_knn<<<4, 256, 0, stream>>>(knn);

  // -- encoder --
  conv1_wmma<<<8192 / 4, 128, 0, stream>>>(xs, whE1, eb1, h1, 8192);    // 4*2*1024
  conv_wmma_relu<32, 64, false, false><<<4096 / 4, 128, 0, stream>>>(
      h1, whE2, eb2, t2, 128, 2 * 512, 4096);                            // 4*2*512
  avgpool2_nhwc<<<4096, 256, 0, stream>>>(t2, h2, 64, 64, 4 * 4096 * 64);
  conv_wmma_relu<64, 32, true, false><<<512 / 4, 128, 0, stream>>>(
      h2, whE3, eb3, t3, 64, 128, 512);                                  // 4*1*128
  avgpool2_cmajor<<<512, 256, 0, stream>>>(t3, h3);

  // -- 47-frame scan: gather-mean then 128x1024x1024 WMMA GEMM --
  h16* bufs[2] = {xpA, xpB};
  for (int f = 0; f < FGEN; ++f) {
    const h16* src = (f == 0) ? h3 : bufs[(f - 1) & 1];
    h16* dst = bufs[f & 1];
    knn_gather<<<512, 256, 0, stream>>>(src, knn, dst);
    gemm_frame_wmma<<<32, 128, 0, stream>>>(dst, kdT, ybuf, f);
  }

  // -- decoder, chunked over the 188 frame-images --
  for (int c0 = 0; c0 < 188; c0 += CHUNK) {
    const h16* yin = ybuf + (size_t)c0 * NPOS * 32;
    int t1n = CHUNK * 2 * 128;    // dec1: Cout=64 @64x64, fused upsample of 32-res y
    conv_wmma_relu<32, 64, true, true><<<t1n / 4, 128, 0, stream>>>(
        yin, whD1, db1, a1, 64, 2 * 128, t1n);
    int t2n = CHUNK * 128;        // dec2: Cout=32 @64x64, zero pad
    conv_wmma_relu<64, 32, false, false><<<t2n / 4, 128, 0, stream>>>(
        a1, whD2, db2, a2, 64, 128, t2n);
    int t3n = CHUNK * 512;        // dec3: Cout=3 @128x128, fused upsample, tanh -> d_out
    conv_dec3_wmma<<<t3n / 4, 128, 0, stream>>>(a2, whD3, db3, out, c0, t3n);
  }
  (void)in_sizes; (void)n_in; (void)out_size; (void)ws_size;
}
